// NetVLAD_65317862637609
// MI455X (gfx1250) — matrix operand order repaired and checked by reading, not compiled, not run on hardware
//
#include <hip/hip_runtime.h>
#include <math.h>

// ---------------- problem constants ----------------
#define NN 32
#define CC 512
#define HH 40
#define WW 40
#define LL (HH * WW)   // 1600 pixels per image
#define KK 64
#define EPSN 1e-12f

// ---------------- WMMA vector types (CDNA5 wave32) ----------------
typedef __attribute__((ext_vector_type(8)))  _Float16 half8;
typedef __attribute__((ext_vector_type(16))) _Float16 v16h;
typedef __attribute__((ext_vector_type(8)))  float    v8f;

union Frag16 { v16h v; half8 h[2]; };

// ---------------- workspace layout (bytes, 256-aligned) ----------------
// All intermediates total ~60 MB -> L2-resident on MI455X (192 MB L2).
#define WS_XN_CL   ((size_t)0)                 // [N][C][L] f16  52,428,800
#define WS_A       ((size_t)52428800)          // [N][K][L] f16   6,553,600
#define WS_W16     ((size_t)58982400)          // [K][C]   f16       65,536
#define WS_ASUM    ((size_t)59047936)          // [N][K]   f32        8,192
#define WS_VLAD    ((size_t)59056128)          // [N][K][C] f32   4,194,304
#define WS_RN2     ((size_t)63250432)          // [N][K]   f32        8,192

// LDS tile row stride for k_assign (padding: 16B-aligned rows, ~2-way max
// bank conflict on ds_load_b128 fragment reads)
#define XT_STRIDE (CC + 8)                     // 520 halves = 1040 B

// =====================================================================
// Kernel 0: convert conv_w -> f16, zero Asum accumulators
// =====================================================================
__global__ __launch_bounds__(256) void k_prep(const float* __restrict__ conv_w,
                                              _Float16* __restrict__ w16,
                                              float* __restrict__ Asum) {
  int t = blockIdx.x * 256 + threadIdx.x;
  if (t < KK * CC) w16[t] = (_Float16)conv_w[t];
  if (t < NN * KK) Asum[t] = 0.0f;
}

// =====================================================================
// Kernel 1: per-pixel L2 normalize over C; emit xn_cl [N][C][L] f16
// (native channel-major layout only). One block = 32 pixels, 256 thr.
// Second pass re-reads x from L2 (x is 105 MB, L2 is 192 MB).
// =====================================================================
__global__ __launch_bounds__(256) void k_normalize(const float* __restrict__ x,
                                                   _Float16* __restrict__ xn_cl) {
  const int blocksPerImg = LL / 32;            // 50
  int n  = blockIdx.x / blocksPerImg;
  int l0 = (blockIdx.x % blocksPerImg) * 32;
  const float* xb = x + (size_t)n * CC * LL;

  int t  = threadIdx.x;
  int ll = t & 31;          // pixel within tile
  int cs = t >> 5;          // channel slice 0..7

  // ---- pass 1: sum of squares per pixel (coalesced over L) ----
  float ss = 0.0f;
  for (int c = cs; c < CC; c += 8) {
    float v = xb[(size_t)c * LL + l0 + ll];
    ss += v * v;
  }
  __shared__ float part[256];
  __shared__ float scale[32];
  part[t] = ss;
  __syncthreads();
  if (t < 32) {
    float s = 0.0f;
#pragma unroll
    for (int i = 0; i < 8; ++i) s += part[t + 32 * i];
    scale[t] = 1.0f / fmaxf(sqrtf(s), EPSN);
  }
  __syncthreads();

  // ---- pass 2: scale + convert, channel-major write ----
  float sc = scale[ll];
  for (int c = cs; c < CC; c += 8) {
    float v = xb[(size_t)c * LL + l0 + ll];
    xn_cl[((size_t)n * CC + c) * LL + l0 + ll] = (_Float16)(v * sc);
  }
}

// =====================================================================
// Kernel 2: GEMM1 (logits^T = conv_w * xn^T) + bias + softmax over K
//           + write A [N][K][L] f16 + Asum partial reduction.
// Block = 128 thr (4 waves). Block tile: 16 pixels x 64 clusters.
// xn tile staged/transposed through LDS (source is channel-major);
// B-fragments come from LDS via ds_load_b128 pairs.
// =====================================================================
__global__ __launch_bounds__(128) void k_assign(const _Float16* __restrict__ xn_cl,
                                                const _Float16* __restrict__ w16,
                                                const float* __restrict__ conv_b,
                                                _Float16* __restrict__ A,
                                                float* __restrict__ Asum) {
  const int tilesPerImg = LL / 16;             // 100
  int n   = blockIdx.x / tilesPerImg;
  int l0  = (blockIdx.x % tilesPerImg) * 16;
  int wv   = threadIdx.x >> 5;                 // wave 0..3
  int lane = threadIdx.x & 31;
  int half = lane >> 4;                        // 0 | 1
  int l15  = lane & 15;
  int kbase = wv * 16;

  // ---- stage + transpose the 16-pixel tile: xt[pixel][channel] ----
  __shared__ _Float16 xt[16][XT_STRIDE];       // 16,640 B
  {
    int tp = threadIdx.x & 15;                 // pixel
    int tc = threadIdx.x >> 4;                 // channel phase 0..7
    for (int c = tc; c < CC; c += 8)
      xt[tp][c] = xn_cl[((size_t)n * CC + c) * LL + l0 + tp];
  }
  __syncthreads();

  v8f acc = {0.f, 0.f, 0.f, 0.f, 0.f, 0.f, 0.f, 0.f};

  // A-fragment source: conv_w row (cluster), K-dim = channels
  const _Float16* arow = w16 + (size_t)(kbase + l15) * CC + (half << 3);
  // B-fragment source: LDS row (pixel), channels off..off+7, off+16..off+23
  const _Float16* brow = &xt[l15][0] + (half << 3);

#pragma unroll 4
  for (int c = 0; c < CC; c += 32) {
    Frag16 a;
    a.h[0] = *(const half8*)(arow + c);        // K = off .. off+7
    a.h[1] = *(const half8*)(arow + c + 16);   // K = off+16 .. off+23
    Frag16 b;                                  // K = off..off+7, off+16..off+23 from LDS
    b.h[0] = *(const half8*)(brow + c);
    b.h[1] = *(const half8*)(brow + c + 16);
    acc = __builtin_amdgcn_wmma_f32_16x16x32_f16(
        false, a.v, false, b.v, (short)0, acc, false, false);
  }

  // ---- bias + softmax over 64 clusters, per pixel (N = lane&15) ----
  float lg[8];
#pragma unroll
  for (int r = 0; r < 8; ++r)
    lg[r] = acc[r] + conv_b[kbase + r + (half << 3)];

  float m = lg[0];
#pragma unroll
  for (int r = 1; r < 8; ++r) m = fmaxf(m, lg[r]);
  m = fmaxf(m, __shfl_xor(m, 16, 32));         // other 8 clusters, same pixel

  __shared__ float redmax[4][16];
  __shared__ float redsum[4][16];
  if (half == 0) redmax[wv][l15] = m;
  __syncthreads();
  float gm = fmaxf(fmaxf(redmax[0][l15], redmax[1][l15]),
                   fmaxf(redmax[2][l15], redmax[3][l15]));

  float e[8], s = 0.0f;
#pragma unroll
  for (int r = 0; r < 8; ++r) { e[r] = __expf(lg[r] - gm); s += e[r]; }
  s += __shfl_xor(s, 16, 32);
  if (half == 0) redsum[wv][l15] = s;
  __syncthreads();
  float inv = 1.0f / (redsum[0][l15] + redsum[1][l15] +
                      redsum[2][l15] + redsum[3][l15]);

#pragma unroll
  for (int r = 0; r < 8; ++r) {
    float av = e[r] * inv;
    int kk = kbase + r + (half << 3);
    A[((size_t)n * KK + kk) * LL + l0 + l15] = (_Float16)av;
    // per-cluster partial sum over the 16 pixels of this tile
    float sp = av;
    sp += __shfl_xor(sp, 1, 32);
    sp += __shfl_xor(sp, 2, 32);
    sp += __shfl_xor(sp, 4, 32);
    sp += __shfl_xor(sp, 8, 32);
    if (l15 == 0) atomicAdd(&Asum[n * KK + kk], sp);
  }
}

// =====================================================================
// Kernel 3: GEMM2 (vlad = A[KxL] * xn_cl^T[LxC]) - Asum*centroids.
// Block = 128 thr (4 waves). Block tile: 64 clusters x 16 channels.
// Wave w: clusters 16w..16w+15, loop all 1600 pixels (50 WMMA steps).
// Both operand streams are L2-resident f16.
// =====================================================================
__global__ __launch_bounds__(128) void k_vlad(const _Float16* __restrict__ A,
                                              const _Float16* __restrict__ xn_cl,
                                              const float* __restrict__ Asum,
                                              const float* __restrict__ centroids,
                                              float* __restrict__ vlad) {
  int n  = blockIdx.x >> 5;                    // / 32
  int ct = blockIdx.x & 31;                    // channel tile
  int wv   = threadIdx.x >> 5;
  int lane = threadIdx.x & 31;
  int half = lane >> 4;
  int l15  = lane & 15;
  int kbase = wv * 16;

  v8f acc = {0.f, 0.f, 0.f, 0.f, 0.f, 0.f, 0.f, 0.f};

  // A-fragment: row = cluster, K-dim = pixels
  const _Float16* arow = A + ((size_t)n * KK + kbase + l15) * LL + (half << 4);
  // B-fragment: column = channel, 16 contiguous pixels (native layout)
  const _Float16* brow = xn_cl + ((size_t)n * CC + ct * 16 + l15) * LL + (half << 4);

#pragma unroll 2
  for (int ls = 0; ls < LL; ls += 32) {
    Frag16 a;
    a.h[0] = *(const half8*)(arow + ls);
    a.h[1] = *(const half8*)(arow + ls + 8);
    v16h b = *(const v16h*)(brow + ls);
    acc = __builtin_amdgcn_wmma_f32_16x16x32_f16(
        false, a.v, false, b, (short)0, acc, false, false);
  }

  // ---- subtract centroid term, store vlad fp32 ----
  int c = ct * 16 + l15;
#pragma unroll
  for (int r = 0; r < 8; ++r) {
    int kk = kbase + r + (half << 3);
    float v = acc[r] - Asum[n * KK + kk] * centroids[(size_t)kk * CC + c];
    vlad[((size_t)n * KK + kk) * CC + c] = v;
  }
}

// =====================================================================
// Kernel 4: intra-normalization over C per (n,k); emit row norm^2 of
// the normalized row for the global norm.
// =====================================================================
__global__ __launch_bounds__(128) void k_intra(const float* __restrict__ vlad,
                                               float* __restrict__ out,
                                               float* __restrict__ rn2) {
  const float* row = vlad + (size_t)blockIdx.x * CC;
  float* orow      = out  + (size_t)blockIdx.x * CC;
  int t = threadIdx.x;

  float s = 0.0f;
  for (int i = t; i < CC; i += 128) { float v = row[i]; s += v * v; }
#pragma unroll
  for (int m = 1; m < 32; m <<= 1) s += __shfl_xor(s, m, 32);
  __shared__ float wsum[4];
  if ((t & 31) == 0) wsum[t >> 5] = s;
  __syncthreads();
  float tot = wsum[0] + wsum[1] + wsum[2] + wsum[3];
  float sc = 1.0f / fmaxf(sqrtf(tot), EPSN);
  for (int i = t; i < CC; i += 128) orow[i] = row[i] * sc;
  if (t == 0) rn2[blockIdx.x] = tot * sc * sc;
}

// =====================================================================
// Kernel 5: global L2 normalization per image (in-place on d_out).
// =====================================================================
__global__ __launch_bounds__(256) void k_gnorm(float* __restrict__ out,
                                               const float* __restrict__ rn2) {
  __shared__ float gsc;
  if (threadIdx.x == 0) {
    float s = 0.0f;
    for (int i = 0; i < KK; ++i) s += rn2[blockIdx.x * KK + i];
    gsc = 1.0f / fmaxf(sqrtf(s), EPSN);
  }
  __syncthreads();
  float g = gsc;
  size_t base = (size_t)blockIdx.x * KK * CC;
  for (int i = threadIdx.x; i < KK * CC; i += 256) out[base + i] *= g;
}

// =====================================================================
extern "C" void kernel_launch(void* const* d_in, const int* in_sizes, int n_in,
                              void* d_out, int out_size, void* d_ws, size_t ws_size,
                              hipStream_t stream) {
  (void)in_sizes; (void)n_in; (void)out_size; (void)ws_size;
  const float* x         = (const float*)d_in[0];   // [N,C,H,W]
  const float* conv_w    = (const float*)d_in[1];   // [K,C]
  const float* conv_b    = (const float*)d_in[2];   // [K]
  const float* centroids = (const float*)d_in[3];   // [K,C]
  float* out = (float*)d_out;

  char* ws = (char*)d_ws;
  _Float16* xn_cl = (_Float16*)(ws + WS_XN_CL);
  _Float16* Abuf  = (_Float16*)(ws + WS_A);
  _Float16* w16   = (_Float16*)(ws + WS_W16);
  float*    Asum  = (float*)(ws + WS_ASUM);
  float*    vlad  = (float*)(ws + WS_VLAD);
  float*    rn2   = (float*)(ws + WS_RN2);

  k_prep<<<(KK * CC + 255) / 256, 256, 0, stream>>>(conv_w, w16, Asum);
  k_normalize<<<NN * (LL / 32), 256, 0, stream>>>(x, xn_cl);
  k_assign<<<NN * (LL / 16), 128, 0, stream>>>(xn_cl, w16, conv_b, Abuf, Asum);
  k_vlad<<<NN * (CC / 16), 128, 0, stream>>>(Abuf, xn_cl, Asum, centroids, vlad);
  k_intra<<<NN * KK, 128, 0, stream>>>(vlad, out, rn2);
  k_gnorm<<<NN, 256, 0, stream>>>(out, rn2);
}
